// VarLayer_79336635892633
// MI455X (gfx1250) — matrix operand order repaired and checked by reading, not compiled, not run on hardware
//
#include <hip/hip_runtime.h>
#include <hip/hip_bf16.h>
#include <math.h>

typedef __attribute__((ext_vector_type(16))) _Float16 v16h;
typedef __attribute__((ext_vector_type(8)))  float    v8f;

#define HW_ 576
#define W24 24
#define N_  8
#define COLT 288                 // (8*576)/16 column tiles, exact
#define CTW  4                   // column tiles per wave (A-fragment reuse)

// ---------------------------------------------------------------------------
// Pack conv weights [Cout][Cin][3][3] (f32) into WMMA A-fragments (f16),
// laid out [mt*Ksteps+kb][lane][16] per the CDNA5 16-bit 16x32 A layout.
// ---------------------------------------------------------------------------
__global__ void pack_wgt(const float* __restrict__ wgt, _Float16* __restrict__ out,
                         int Cin, int rowTiles, int Ksteps)
{
    const int idx = blockIdx.x * blockDim.x + threadIdx.x;
    const int total = rowTiles * Ksteps * 512;
    if (idx >= total) return;
    const int e    = idx & 15;
    const int lane = (idx >> 4) & 31;
    const int frag = idx >> 9;
    const int kb = frag % Ksteps;
    const int mt = frag / Ksteps;
    const int v = e >> 1, j = e & 1;
    const int hi = (lane >= 16);
    const int base = (v < 4) ? (hi ? 8 : 0) : (hi ? 24 : 16);
    const int k = kb * 32 + base + 2 * (v & 3) + j;
    const int K = Cin * 9;
    float x = 0.0f;
    if (k < K) {
        const int m = mt * 16 + (lane & 15);
        x = wgt[(m * Cin + k / 9) * 9 + (k % 9)];
    }
    out[idx] = (_Float16)x;
}

// ---------------------------------------------------------------------------
// Pack im2col patches of a [N,Cin,24,24] f32 tensor into WMMA B-fragments
// (f16), laid out [ct*Ksteps+kb][lane][16] per the 32x16 B layout.
// ---------------------------------------------------------------------------
__global__ void pack_im2col(const float* __restrict__ in, _Float16* __restrict__ out,
                            int Cin, int Ksteps)
{
    const int idx = blockIdx.x * blockDim.x + threadIdx.x;
    const int total = COLT * Ksteps * 512;
    if (idx >= total) return;
    const int e    = idx & 15;
    const int lane = (idx >> 4) & 31;
    const int frag = idx >> 9;
    const int kb = frag % Ksteps;
    const int ct = frag / Ksteps;
    const int v = e >> 1, j = e & 1;
    const int k = kb * 32 + ((v < 4) ? 0 : 16) + ((lane >= 16) ? 8 : 0) + 2 * (v & 3) + j;
    const int col = ct * 16 + (lane & 15);
    const int K = Cin * 9;
    float x = 0.0f;
    if (k < K) {
        const int n   = col / HW_;
        const int pix = col % HW_;
        const int y0 = pix / W24, x0 = pix % W24;
        const int c  = k / 9,  rs = k % 9;
        const int yy = y0 + rs / 3 - 1;
        const int xx = x0 + rs % 3 - 1;
        if (yy >= 0 && yy < 24 && xx >= 0 && xx < 24)
            x = in[((n * Cin + c) * 24 + yy) * 24 + xx];
    }
    out[idx] = (_Float16)x;
}

// ---------------------------------------------------------------------------
// Packed-fragment implicit GEMM: per wave, one 16(Cout) x 64(pixel) tile:
// 4 column tiles share one A fragment -> per K-step, 1 A load + 4 B loads
// feed 4 independent v_wmma_f32_16x16x32_f16 (40 B/wmma vs 64 B/wmma).
// KSTEPS compile-time -> fully unrolled. act: 0=none, 1=leaky, 2=sigmoid
// ---------------------------------------------------------------------------
template<int KSTEPS>
__global__ void conv_wmma_packed(const v16h* __restrict__ apack,
                                 const v16h* __restrict__ bpack,
                                 const float* __restrict__ bias,
                                 float* __restrict__ out,
                                 int Cout, int act)
{
    const int lane = threadIdx.x & 31;
    const int wave = threadIdx.x >> 5;
    const int rowTiles  = Cout >> 4;
    const int colGroups = COLT / CTW;        // 72
    const int tile = blockIdx.x * (blockDim.x >> 5) + wave;
    if (tile >= rowTiles * colGroups) return;   // wave-uniform exit, EXEC full

    const int mt  = tile / colGroups;
    const int ct0 = (tile % colGroups) * CTW;
    const v16h* ap = apack + (size_t)(mt * KSTEPS) * 32 + lane;
    const v16h* bp0 = bpack + (size_t)(ct0 * KSTEPS) * 32 + lane;

    // warm next group's B panel in GL2 (speculative, no counters)
    __builtin_prefetch(bp0 + (size_t)(CTW * KSTEPS) * 32, 0, 0);

    v8f acc0 = {}, acc1 = {}, acc2 = {}, acc3 = {};
    #pragma unroll
    for (int kb = 0; kb < KSTEPS; ++kb) {
        const v16h a  = ap[(size_t)kb * 32];
        const v16h b0 = bp0[(size_t)(0 * KSTEPS + kb) * 32];
        const v16h b1 = bp0[(size_t)(1 * KSTEPS + kb) * 32];
        const v16h b2 = bp0[(size_t)(2 * KSTEPS + kb) * 32];
        const v16h b3 = bp0[(size_t)(3 * KSTEPS + kb) * 32];
        acc0 = __builtin_amdgcn_wmma_f32_16x16x32_f16(false, a, false, b0, (short)0, acc0, false, false);
        acc1 = __builtin_amdgcn_wmma_f32_16x16x32_f16(false, a, false, b1, (short)0, acc1, false, false);
        acc2 = __builtin_amdgcn_wmma_f32_16x16x32_f16(false, a, false, b2, (short)0, acc2, false, false);
        acc3 = __builtin_amdgcn_wmma_f32_16x16x32_f16(false, a, false, b3, (short)0, acc3, false, false);
    }

    const int lrow = lane & 15;
    const int hi   = (lane >= 16);
    v8f accs[CTW] = {acc0, acc1, acc2, acc3};
    #pragma unroll
    for (int c = 0; c < CTW; ++c) {
        const int col = (ct0 + c) * 16 + lrow;   // always < 4608
        const int n   = col / HW_;
        const int pix = col % HW_;
        #pragma unroll
        for (int r = 0; r < 8; ++r) {
            const int m = mt * 16 + (hi ? (8 + r) : r);
            float v = accs[c][r] + bias[m];
            if (act == 1)      v = (v >= 0.0f) ? v : 0.01f * v;
            else if (act == 2) v = 1.0f / (1.0f + __expf(-v));
            out[(n * Cout + m) * HW_ + pix] = v;
        }
    }
}

// ---------------------------------------------------------------------------
// SE path: global avg-pool -> 1x1 (64->32) leaky -> 1x1 (32->4) sigmoid
// ---------------------------------------------------------------------------
__global__ void se_kernel(const float* __restrict__ x,
                          const float* __restrict__ sw1, const float* __restrict__ sb1,
                          const float* __restrict__ sw2, const float* __restrict__ sb2,
                          float* __restrict__ se)
{
    __shared__ float p[N_ * 64];
    __shared__ float h1[N_ * 32];
    const int t = threadIdx.x;
    for (int idx = t; idx < N_ * 64; idx += blockDim.x) {
        const float* base = x + idx * HW_;
        float s = 0.0f;
        for (int i = 0; i < HW_; ++i) s += base[i];
        p[idx] = s * (1.0f / (float)HW_);
    }
    __syncthreads();
    for (int idx = t; idx < N_ * 32; idx += blockDim.x) {
        const int nn = idx >> 5, o = idx & 31;
        float s = sb1[o];
        for (int c = 0; c < 64; ++c) s += sw1[o * 64 + c] * p[nn * 64 + c];
        h1[idx] = (s >= 0.0f) ? s : 0.01f * s;
    }
    __syncthreads();
    for (int idx = t; idx < N_ * 4; idx += blockDim.x) {
        const int nn = idx >> 2, o = idx & 3;
        float s = sb2[o];
        for (int c = 0; c < 32; ++c) s += sw2[o * 32 + c] * h1[nn * 32 + c];
        se[idx] = 1.0f / (1.0f + __expf(-s));
    }
}

// ---------------------------------------------------------------------------
// Implicit-operator CG for the normal equations (weighted-Laplacian stencil).
// ---------------------------------------------------------------------------
__device__ __forceinline__ float block_dot(const float* a, const float* b, float* red)
{
    const int t = threadIdx.x;
    float s = 0.0f;
    for (int i = t; i < HW_; i += 256) s += a[i] * b[i];
    red[t] = s;
    __syncthreads();
    for (int k = 128; k > 0; k >>= 1) {
        if (t < k) red[t] += red[t + k];
        __syncthreads();
    }
    const float r = red[0];
    __syncthreads();
    return r;
}

__device__ __forceinline__ void apply_ata(const float* w, const float* pv, float* ap)
{
    const int t = threadIdx.x;
    for (int i = t; i < HW_; i += 256) ap[i] = 1e-12f * pv[i];
    __syncthreads();
    for (int idx = t; idx < 4 * HW_; idx += 256) {
        const int k = idx / HW_;
        const int i = idx - k * HW_;
        const float wt = w[idx];
        const float w2 = wt * wt;
        int off = 0; bool on = false;
        if (k == 0)      { off = 1;  on = ((i + 1) % W24 != 0) && (i + 1 < HW_); }
        else if (k == 1) { off = 24; on = (i + 24 < HW_); }
        else if (k == 2) { off = 2;  on = ((i + 2) % W24 != 0) && ((i + 1) % W24 != 0) && (i + 2 < HW_); }
        else             { off = 48; on = (i + 48 < HW_); }
        if (i == HW_ - 1) {
            atomicAdd(&ap[i], w2 * pv[i]);
        } else if (on) {
            const float tv = w2 * (pv[i] - pv[i + off]);
            atomicAdd(&ap[i],       tv);
            atomicAdd(&ap[i + off], -tv);
        }
    }
    __syncthreads();
}

__global__ void cg_solve(const float* __restrict__ att,
                         const float* __restrict__ grad,
                         float* __restrict__ sol)
{
    const int b = blockIdx.x;            // 0..31 = n*4 + g
    const int nn = b >> 2, g = b & 3;
    __shared__ float w[4 * HW_];
    __shared__ float rhs[HW_], xv[HW_], rv[HW_], pv[HW_], ap[HW_];
    __shared__ float red[256];
    const int t = threadIdx.x;

    for (int idx = t; idx < 4 * HW_; idx += 256) {
        const int k = idx / HW_, i = idx - (idx / HW_) * HW_;
        w[idx] = att[((nn * 16) + (g * 4 + k)) * HW_ + i];
    }
    for (int i = t; i < HW_; i += 256) { rhs[i] = 0.0f; xv[i] = 0.0f; }
    __syncthreads();

    for (int idx = t; idx < 4 * HW_; idx += 256) {
        const int k = idx / HW_;
        const int i = idx - k * HW_;
        const float wt = w[idx];
        const float gv = grad[((nn * 16) + (g * 4 + k)) * HW_ + i];
        const float c = wt * wt * gv;
        int off = 0; bool on = false;
        if (k == 0)      { off = 1;  on = ((i + 1) % W24 != 0) && (i + 1 < HW_); }
        else if (k == 1) { off = 24; on = (i + 24 < HW_); }
        else if (k == 2) { off = 2;  on = ((i + 2) % W24 != 0) && ((i + 1) % W24 != 0) && (i + 2 < HW_); }
        else             { off = 48; on = (i + 48 < HW_); }
        if (i == HW_ - 1) {
            atomicAdd(&rhs[i], c);
        } else if (on) {
            atomicAdd(&rhs[i],       c);
            atomicAdd(&rhs[i + off], -c);
        }
    }
    __syncthreads();

    for (int i = t; i < HW_; i += 256) { rv[i] = rhs[i]; pv[i] = rhs[i]; }
    __syncthreads();
    float rs = block_dot(rv, rv, red);

    for (int it = 0; it < 96; ++it) {
        apply_ata(w, pv, ap);
        const float pap   = block_dot(pv, ap, red);
        const float alpha = rs / fmaxf(pap, 1e-30f);
        for (int i = t; i < HW_; i += 256) {
            xv[i] += alpha * pv[i];
            rv[i] -= alpha * ap[i];
        }
        __syncthreads();
        const float rs2  = block_dot(rv, rv, red);
        const float beta = rs2 / fmaxf(rs, 1e-30f);
        for (int i = t; i < HW_; i += 256) pv[i] = rv[i] + beta * pv[i];
        __syncthreads();
        rs = rs2;
    }
    for (int i = t; i < HW_; i += 256) sol[b * HW_ + i] = xv[i];
}

// ---------------------------------------------------------------------------
// GroupNorm(1 group) per sample + se gating -> pw-conv input
// ---------------------------------------------------------------------------
__global__ void gn_kernel(const float* __restrict__ sol,
                          const float* __restrict__ gnw, const float* __restrict__ gnb,
                          const float* __restrict__ se, float* __restrict__ pin)
{
    const int nn = blockIdx.x;
    const int t = threadIdx.x;
    __shared__ float r1[256], r2[256];
    float s = 0.0f, s2 = 0.0f;
    for (int i = t; i < 4 * HW_; i += 256) {
        const float v = sol[nn * 4 * HW_ + i];
        s += v; s2 += v * v;
    }
    r1[t] = s; r2[t] = s2;
    __syncthreads();
    for (int k = 128; k > 0; k >>= 1) {
        if (t < k) { r1[t] += r1[t + k]; r2[t] += r2[t + k]; }
        __syncthreads();
    }
    const float mu  = r1[0] * (1.0f / (4.0f * HW_));
    const float var = r2[0] * (1.0f / (4.0f * HW_)) - mu * mu;
    const float inv = rsqrtf(var + 1e-5f);
    for (int i = t; i < 4 * HW_; i += 256) {
        const int g = i / HW_;
        const float v = (sol[nn * 4 * HW_ + i] - mu) * inv * gnw[g] + gnb[g];
        pin[nn * 4 * HW_ + i] = v * se[nn * 4 + g];
    }
}

// ---------------------------------------------------------------------------
extern "C" void kernel_launch(void* const* d_in, const int* in_sizes, int n_in,
                              void* d_out, int out_size, void* d_ws, size_t ws_size,
                              hipStream_t stream)
{
    const float* x   = (const float*)d_in[0];
    const float* gw1 = (const float*)d_in[2];
    const float* gb1 = (const float*)d_in[3];
    const float* gw2 = (const float*)d_in[4];
    const float* gb2 = (const float*)d_in[5];
    const float* aw1 = (const float*)d_in[6];
    const float* ab1 = (const float*)d_in[7];
    const float* aw2 = (const float*)d_in[8];
    const float* ab2 = (const float*)d_in[9];
    const float* sw1 = (const float*)d_in[10];
    const float* sb1 = (const float*)d_in[11];
    const float* sw2 = (const float*)d_in[12];
    const float* sb2 = (const float*)d_in[13];
    const float* gnw = (const float*)d_in[14];
    const float* gnb = (const float*)d_in[15];
    const float* pw  = (const float*)d_in[16];
    const float* pb  = (const float*)d_in[17];
    float* out = (float*)d_out;

    // ---- workspace layout (float units; every region 32B-aligned) ----
    float* ws   = (float*)d_ws;
    float* t1g  = ws;                        // 8*64*576           = 294912
    float* t1a  = t1g + 294912;              // 294912
    float* grad = t1a + 294912;              // 8*16*576           = 73728
    float* att  = grad + 73728;              // 73728
    float* sev  = att + 73728;               // 32
    float* sol  = sev + 32;                  // 32*576             = 18432
    float* pin  = sol + 18432;               // 8*4*576            = 18432
    float* f    = pin + 18432;               // start of f16 regions
    _Float16* xpack   = (_Float16*)f;                  // 288*18*512 = 2654208 halfs
    _Float16* gpackB  = xpack  + 2654208;              // 2654208
    _Float16* apackB  = gpackB + 2654208;              // 2654208
    _Float16* pinpack = apackB + 2654208;              // 288*2*512 = 294912
    _Float16* gw1p    = pinpack + 294912;              // 4*18*512 = 36864
    _Float16* aw1p    = gw1p + 36864;                  // 36864
    _Float16* gw2p    = aw1p + 36864;                  // 1*18*512 = 9216
    _Float16* aw2p    = gw2p + 9216;                   // 9216
    _Float16* pwp     = aw2p + 9216;                   // 8*2*512 = 8192

    auto cdiv = [](int a, int b) { return (a + b - 1) / b; };
    auto conv_blocks = [&](int Cout) { return cdiv((Cout >> 4) * (COLT / CTW), 8); };

    // ---- weight packs (tiny) ----
    pack_wgt<<<cdiv(4 * 18 * 512, 256), 256, 0, stream>>>(gw1, gw1p, 64, 4, 18);
    pack_wgt<<<cdiv(4 * 18 * 512, 256), 256, 0, stream>>>(aw1, aw1p, 64, 4, 18);
    pack_wgt<<<cdiv(1 * 18 * 512, 256), 256, 0, stream>>>(gw2, gw2p, 64, 1, 18);
    pack_wgt<<<cdiv(1 * 18 * 512, 256), 256, 0, stream>>>(aw2, aw2p, 64, 1, 18);
    pack_wgt<<<cdiv(8 *  2 * 512, 256), 256, 0, stream>>>(pw,  pwp,   4, 8,  2);

    // ---- layer 1: shared im2col of x, two WMMA GEMMs ----
    pack_im2col<<<cdiv(COLT * 18 * 512, 256), 256, 0, stream>>>(x, xpack, 64, 18);
    conv_wmma_packed<18><<<conv_blocks(64), 256, 0, stream>>>((const v16h*)gw1p, (const v16h*)xpack, gb1, t1g, 64, 1);
    conv_wmma_packed<18><<<conv_blocks(64), 256, 0, stream>>>((const v16h*)aw1p, (const v16h*)xpack, ab1, t1a, 64, 1);

    // ---- layer 2: per-branch im2col + WMMA GEMMs ----
    pack_im2col<<<cdiv(COLT * 18 * 512, 256), 256, 0, stream>>>(t1g, gpackB, 64, 18);
    pack_im2col<<<cdiv(COLT * 18 * 512, 256), 256, 0, stream>>>(t1a, apackB, 64, 18);
    conv_wmma_packed<18><<<conv_blocks(16), 256, 0, stream>>>((const v16h*)gw2p, (const v16h*)gpackB, gb2, grad, 16, 0);
    conv_wmma_packed<18><<<conv_blocks(16), 256, 0, stream>>>((const v16h*)aw2p, (const v16h*)apackB, ab2, att,  16, 2);

    // ---- SE, CG solve, GroupNorm ----
    se_kernel<<<1, 256, 0, stream>>>(x, sw1, sb1, sw2, sb2, sev);
    cg_solve<<<32, 256, 0, stream>>>(att, grad, sol);
    gn_kernel<<<N_, 256, 0, stream>>>(sol, gnw, gnb, sev, pin);

    // ---- final 4->128 conv ----
    pack_im2col<<<cdiv(COLT * 2 * 512, 256), 256, 0, stream>>>(pin, pinpack, 4, 2);
    conv_wmma_packed<2><<<conv_blocks(128), 256, 0, stream>>>((const v16h*)pwp, (const v16h*)pinpack, pb, out, 128, 0);
}
